// ScaledDotProductAttention_52527450030613
// MI455X (gfx1250) — compile-verified, matrix operand order checked by reading, not run on hardware
//
#include <hip/hip_runtime.h>

typedef _Float16 v16h __attribute__((ext_vector_type(16)));
typedef _Float16 v8h  __attribute__((ext_vector_type(8)));
typedef _Float16 v4h  __attribute__((ext_vector_type(4)));
typedef float    v8f  __attribute__((ext_vector_type(8)));
typedef float    v4f  __attribute__((ext_vector_type(4)));

#define S_LEN  4096
#define D_DIM  64
#define TQ     64          // query rows per block (4 waves x 16)
#define TK     128         // KV tile
#define KSTR   72          // K-tile LDS row stride (halves), 16B aligned
#define VSTR   136         // Vt / P LDS row stride (halves), 16B aligned

// Build a v16h WMMA A/B fragment from two contiguous 16-byte LDS chunks.
__device__ __forceinline__ v16h make_ab(const _Float16* base, int off_lo, int off_hi) {
  v8h lo = *(const v8h*)(base + off_lo);
  v8h hi = *(const v8h*)(base + off_hi);
  v16h r;
#pragma unroll
  for (int i = 0; i < 8; ++i) { r[i] = lo[i]; r[i + 8] = hi[i]; }
  return r;
}

__global__ __launch_bounds__(128)
void fa_fwd_kernel(const float* __restrict__ Q, const float* __restrict__ K,
                   const float* __restrict__ V, float* __restrict__ O) {
  __shared__ __align__(16) _Float16 Klds[TK * KSTR];        // [key][feat]
  __shared__ __align__(16) _Float16 Vt  [D_DIM * VSTR];     // [feat][key] transposed
  __shared__ __align__(16) _Float16 Plds[4 * 16 * VSTR];    // per-wave P strips

  const int tid  = threadIdx.x;
  const int wave = tid >> 5;
  const int lane = tid & 31;
  const int half = lane >> 4;
  const int l16  = lane & 15;

  const int bh    = blockIdx.y;
  const int qbase = blockIdx.x * TQ;
  const size_t plane = (size_t)bh * S_LEN * D_DIM;
  const float* Qp = Q + plane;
  const float* Kp = K + plane;
  const float* Vp = V + plane;
  float*       Op = O + plane;

  // 1/sqrt(64) * log2(e): scores land directly in log2 domain -> raw v_exp_f32.
  const float qscale = 0.125f * 1.44269504088896340736f;

  // ---- Q A-fragments (rows qbase + 16*wave + l16), vectorized global loads ----
  v16h aq[2];
  {
    const float* qrow = Qp + (size_t)(qbase + 16 * wave + l16) * D_DIM;
#pragma unroll
    for (int c = 0; c < 2; ++c) {
      v4f f0 = *(const v4f*)(qrow + 32 * c + 8 * half);
      v4f f1 = *(const v4f*)(qrow + 32 * c + 8 * half + 4);
      v4f f2 = *(const v4f*)(qrow + 32 * c + 16 + 8 * half);
      v4f f3 = *(const v4f*)(qrow + 32 * c + 16 + 8 * half + 4);
#pragma unroll
      for (int j = 0; j < 4; ++j) {
        aq[c][j]      = (_Float16)(f0[j] * qscale);
        aq[c][j + 4]  = (_Float16)(f1[j] * qscale);
        aq[c][j + 8]  = (_Float16)(f2[j] * qscale);
        aq[c][j + 12] = (_Float16)(f3[j] * qscale);
      }
    }
  }

  v8f sacc[8], oacc[4];
#pragma unroll
  for (int t = 0; t < 4; ++t)
#pragma unroll
    for (int r = 0; r < 8; ++r) oacc[t][r] = 0.0f;

  float rmax[8], rsum[8];
#pragma unroll
  for (int r = 0; r < 8; ++r) { rmax[r] = -1e30f; rsum[r] = 0.0f; }

  _Float16* Pw = Plds + wave * 16 * VSTR;

  for (int kv = 0; kv < S_LEN; kv += TK) {
    __syncthreads();  // previous iteration's LDS reads complete

    // ---- Stage K (row-major f16) and V (transposed f16), float4 global loads ----
#pragma unroll 4
    for (int i = 0; i < 16; ++i) {
      int idx  = tid + 128 * i;       // 0..2047 float4 slots, coalesced over tid
      int row  = idx >> 4;            // key within tile (0..127)
      int col4 = idx & 15;            // float4 column (0..15)
      v4f kf = *(const v4f*)(Kp + (size_t)(kv + row) * D_DIM + col4 * 4);
      v4h kh;
#pragma unroll
      for (int j = 0; j < 4; ++j) kh[j] = (_Float16)kf[j];
      *(v4h*)(Klds + row * KSTR + col4 * 4) = kh;          // ds_store_b64

      v4f vf = *(const v4f*)(Vp + (size_t)(kv + row) * D_DIM + col4 * 4);
#pragma unroll
      for (int j = 0; j < 4; ++j)
        Vt[(col4 * 4 + j) * VSTR + row] = (_Float16)vf[j]; // transpose scatter
    }
    __syncthreads();

    // Prefetch next tile while computing on this one.
    if (kv + TK < S_LEN) {
      __builtin_prefetch(Kp + (size_t)(kv + TK + (tid & 127)) * D_DIM, 0, 0);
      __builtin_prefetch(Vp + (size_t)(kv + TK + (tid & 127)) * D_DIM, 0, 0);
    }

    // ---- S = Q*K^T : 8 n-tiles x 2 k-chunks = 16 WMMA / wave ----
#pragma unroll
    for (int t = 0; t < 8; ++t)
#pragma unroll
      for (int r = 0; r < 8; ++r) sacc[t][r] = 0.0f;

#pragma unroll
    for (int c = 0; c < 2; ++c) {
#pragma unroll
      for (int t = 0; t < 8; ++t) {
        const _Float16* krow = Klds + (16 * t + l16) * KSTR;
        v16h bk = make_ab(krow, 32 * c + 8 * half, 32 * c + 16 + 8 * half);
        sacc[t] = __builtin_amdgcn_wmma_f32_16x16x32_f16(
            false, aq[c], false, bk, (short)0, sacc[t], false, false);
      }
    }

    // ---- Online softmax (log2 domain), rows = r + 8*half ----
#pragma unroll
    for (int r = 0; r < 8; ++r) {
      float m = sacc[0][r];
#pragma unroll
      for (int t = 1; t < 8; ++t) m = fmaxf(m, sacc[t][r]);
#pragma unroll
      for (int off = 8; off >= 1; off >>= 1)
        m = fmaxf(m, __shfl_xor(m, off, 32));       // xor<16 stays inside half
      float nm   = fmaxf(rmax[r], m);
      float corr = __builtin_amdgcn_exp2f(rmax[r] - nm);
      rmax[r] = nm;
      float ps = 0.0f;
#pragma unroll
      for (int t = 0; t < 8; ++t) {
        float p = __builtin_amdgcn_exp2f(sacc[t][r] - nm);
        ps += p;
        Pw[(r + 8 * half) * VSTR + 16 * t + l16] = (_Float16)p;
      }
#pragma unroll
      for (int off = 8; off >= 1; off >>= 1)
        ps += __shfl_xor(ps, off, 32);
      rsum[r] = rsum[r] * corr + ps;
#pragma unroll
      for (int t = 0; t < 4; ++t) oacc[t][r] *= corr;
    }

    // Wave-private P strip; DS is in-order per wave — belt-and-braces wait.
    asm volatile("s_wait_dscnt 0" ::: "memory");

    // ---- O += P*V : 4 key-chunks x 4 d-tiles = 16 WMMA / wave ----
#pragma unroll
    for (int c = 0; c < 4; ++c) {
      v16h ap = make_ab(Pw + l16 * VSTR, 32 * c + 8 * half, 32 * c + 16 + 8 * half);
#pragma unroll
      for (int t = 0; t < 4; ++t) {
        const _Float16* vrow = Vt + (16 * t + l16) * VSTR;  // row = feature d
        v16h bv = make_ab(vrow, 32 * c + 8 * half, 32 * c + 16 + 8 * half);
        oacc[t] = __builtin_amdgcn_wmma_f32_16x16x32_f16(
            false, ap, false, bv, (short)0, oacc[t], false, false);
      }
    }
  }

  // ---- Normalize and store (fp32, coalesced across l16) ----
#pragma unroll
  for (int r = 0; r < 8; ++r) {
    float inv = __builtin_amdgcn_rcpf(rsum[r]);
    int row = qbase + 16 * wave + r + 8 * half;
    float* orow = Op + (size_t)row * D_DIM;
#pragma unroll
    for (int t = 0; t < 4; ++t)
      orow[16 * t + l16] = oacc[t][r] * inv;
  }
}

extern "C" void kernel_launch(void* const* d_in, const int* in_sizes, int n_in,
                              void* d_out, int out_size, void* d_ws, size_t ws_size,
                              hipStream_t stream) {
  (void)n_in; (void)out_size; (void)d_ws; (void)ws_size;
  const float* Q = (const float*)d_in[0];
  const float* K = (const float*)d_in[1];
  const float* V = (const float*)d_in[2];
  float* O = (float*)d_out;
  int bh = in_sizes[0] / (S_LEN * D_DIM);   // B*H = 24
  dim3 grid(S_LEN / TQ, bh);
  fa_fwd_kernel<<<grid, 128, 0, stream>>>(Q, K, V, O);
}
